// NLSPNModel_27977416966373
// MI455X (gfx1250) — matrix operand order repaired and checked by reading, not compiled, not run on hardware
//
#include <hip/hip_runtime.h>
#include <cstdint>
#include <cstddef>

// ---------------------------------------------------------------------------
// NLSPN depth propagation for MI455X (gfx1250).
//
// Phase 1 (once): 3x3 conv (8->24ch) on guidance as a 24x72 GEMM per
//   16-pixel tile using v_wmma_f32_16x16x32_f16, + TGASS affinity
//   normalization. Emits compressed per-pixel sampling state:
//     tapYX[j][p] : (int16 y0)<<16 | (int16 x0)      (4 B)
//     tapF [j][p] : half2(fy, fx)                    (4 B)
//     tapA [j][p] : half affinity                    (2 B)
//   => 90 B/pixel = 154 MB total: the 18-step working set (+feat+conf
//   ~14 MB) fits in the 192 MB global L2, so steps 2..18 are L2-resident.
//
// Phase 2 (x18): out[p] = sum_j aff_j * bilinear(g, y_j, x_j) with
//   g = feat*conf maintained across steps (g_{t+1} = conf .* DCN(g_t)),
//   final step writes DCN(g_17) without the conf multiply. Tap state is
//   DMA'd into LDS per block via the Tensor Data Mover (tensor_load_to_lds),
//   synchronized with s_wait_tensorcnt.
// ---------------------------------------------------------------------------

#define BATCH 4
#define HH    352
#define WW    1216
#define NPIX  (BATCH * HH * WW)          // 1,712,128  (divisible by 256)
#define PLANE (HH * WW)                  // 428,032
#define TW    (WW / 16)                  // 76 tiles per row
#define NTILE (BATCH * HH * TW)          // 107,008  (divisible by 8)
#define PROP_STEPS 18
#define KSTR 104                         // padded K stride (halves): 208 B rows,
                                         // 16-B aligned, 52-dword lane stride
                                         // -> distinct LDS bank groups

typedef __attribute__((ext_vector_type(16))) _Float16 v16h;
typedef __attribute__((ext_vector_type(8)))  _Float16 v8h;
typedef __attribute__((ext_vector_type(8)))  float    v8f;
typedef __attribute__((ext_vector_type(4)))  unsigned v4u;
typedef __attribute__((ext_vector_type(8)))  int      v8i;
typedef __attribute__((ext_vector_type(4)))  int      v4i;

union H2U { unsigned u; _Float16 h[2]; };

__device__ __forceinline__ unsigned pack_h2(float lo, float hi) {
    H2U v; v.h[0] = (_Float16)lo; v.h[1] = (_Float16)hi; return v.u;
}

__device__ __forceinline__ v16h cat8(v8h lo, v8h hi) {
    return __builtin_shufflevector(lo, hi, 0, 1, 2, 3, 4, 5, 6, 7,
                                           8, 9, 10, 11, 12, 13, 14, 15);
}

__device__ __forceinline__ void wait_tensorcnt0() {
#if __has_builtin(__builtin_amdgcn_s_wait_tensorcnt)
    __builtin_amdgcn_s_wait_tensorcnt(0);
#else
    asm volatile("s_wait_tensorcnt 0x0" ::: "memory");
#endif
}

// ---------------------------------------------------------------------------
// Tensor Data Mover: 2D tile (tile_w elems x tile_h rows, row stride in
// elements) from global -> LDS. D# per CDNA5 ISA §8.3-8.5. Not in a cluster
// => workgroup_mask must be 0. EXEC is ignored by TDM, so callers must guard
// to a single lane per block.
// ---------------------------------------------------------------------------
#if __has_builtin(__builtin_amdgcn_tensor_load_to_lds)
#define USE_TDM 1
__device__ __forceinline__ void tdm_load_2d(unsigned lds_addr, const void* gptr,
                                            unsigned tile_w, unsigned tile_h,
                                            unsigned long long row_stride_elems,
                                            unsigned data_size_code /*0:1B 1:2B 2:4B*/)
{
    unsigned long long ga = (unsigned long long)(uintptr_t)gptr;
    v4u g0;
    g0[0] = 1u;                                                  // count=1, user D#
    g0[1] = lds_addr;                                            // LDS byte addr
    g0[2] = (unsigned)(ga & 0xFFFFFFFFu);                        // global_addr lo
    g0[3] = (unsigned)((ga >> 32) & 0x01FFFFFFu) | (2u << 30);   // addr hi | type=2

    unsigned long long td0 = 0xFFFFFFFFull;                      // tensor_dim0 (no OOB)
    unsigned long long s0  = row_stride_elems;                   // tensor_dim0_stride
    v8i g1;
    g1[0] = (int)(data_size_code << 16);                         // mask=0 | data_size
    g1[1] = (int)((td0 & 0xFFFFull) << 16);                      // td0 lo16 in [31:16]
    g1[2] = (int)(((td0 >> 16) & 0xFFFFull) |
                  ((unsigned long long)(tile_h & 0xFFFFu) << 16)); // td0 hi | td1 lo
    g1[3] = (int)((tile_w & 0xFFFFu) << 16);                     // td1 hi=0 | tile_dim0
    g1[4] = (int)(tile_h & 0xFFFFu);                             // tile_dim1 | tile_dim2=0
    g1[5] = (int)(s0 & 0xFFFFFFFFull);                           // stride0 lo32
    g1[6] = (int)((s0 >> 32) & 0xFFFFull);                       // stride0 hi | stride1 lo=0
    g1[7] = 0;                                                   // stride1 hi
    v4i z4 = {0, 0, 0, 0};
#if defined(__clang_major__) && __clang_major__ >= 23
    v8i z8 = {0, 0, 0, 0, 0, 0, 0, 0};
    __builtin_amdgcn_tensor_load_to_lds(g0, g1, z4, z4, z8, 0);
#else
    __builtin_amdgcn_tensor_load_to_lds(g0, g1, z4, z4, 0);
#endif
}
#else
#define USE_TDM 0
#endif

// ---------------------------------------------------------------------------
// Kernel 1: guidance conv (WMMA GEMM) + TGASS -> compressed tap state.
// Block = 256 threads = 8 waves; each wave owns one 16-pixel tile (same row).
// GEMM: C[24x16] = W[24x72] * im2col[72x16], padded to 32x96x16,
// done as 2 M-tiles x 3 K-chunks of v_wmma_f32_16x16x32_f16.
// A and B are stored K-contiguous per row so every WMMA operand fragment is
// loaded as contiguous ds_load_b128 pairs (no u16 gathers).
// ---------------------------------------------------------------------------
__global__ void __launch_bounds__(256)
offset_aff_wmma_kernel(const float* __restrict__ guidance,
                       const float* __restrict__ w_oa,
                       const float* __restrict__ b_oa,
                       const float* __restrict__ aff_scale,
                       unsigned* __restrict__ tapYX,
                       unsigned* __restrict__ tapF,
                       _Float16* __restrict__ tapA)
{
    __shared__ _Float16 Apad[32][KSTR];        // padded weights   (6.5 KB)
    __shared__ _Float16 Bbuf[8][16][KSTR];     // im2col, [pix][K] (26 KB)
    __shared__ float    Cbuf[8][24][16];       // GEMM result      (12 KB)

    const int tid  = threadIdx.x;
    const int wave = tid >> 5;
    const int lane = tid & 31;

    // Stage padded A (weights): Apad[m][k] ; w_oa is [24][8*3*3] = [m][k].
    for (int idx = tid; idx < 32 * 96; idx += 256) {
        int m = idx / 96, k = idx % 96;
        Apad[m][k] = (m < 24 && k < 72) ? (_Float16)w_oa[m * 72 + k] : (_Float16)0.0f;
    }

    const int tile = blockIdx.x * 8 + wave;    // NTILE divisible by 8: always valid
    const int b    = tile / (HH * TW);
    const int rem  = tile % (HH * TW);
    const int y    = rem / TW;
    const int xt   = (rem % TW) * 16;

    // im2col: Bbuf[wave][n][k], k = ic*9 + kh*3 + kw, pixel n at (y, xt+n).
    // Read order stays k-major (consecutive lanes -> consecutive gx, coalesced);
    // only the LDS store is transposed.
    for (int idx = lane; idx < 96 * 16; idx += 32) {
        int k = idx >> 4, n = idx & 15;
        float v = 0.0f;
        if (k < 72) {
            int ic = k / 9, r = k % 9, kh = r / 3, kw = r % 3;
            int gy = y + kh - 1, gx = xt + n + kw - 1;
            if (gy >= 0 && gy < HH && gx >= 0 && gx < WW)
                v = guidance[(((size_t)b * 8 + ic) * HH + gy) * (size_t)WW + gx];
        }
        Bbuf[wave][n][k] = (_Float16)v;
    }
    __syncthreads();

    // WMMA fragments per ISA §7.12.2 (16-bit A 16x32, B 32x16, f32 C 16x16).
    // A lane (wave32): M = lane&15; halves 0..7 -> K {0..7}+hiA,
    //                  halves 8..15 -> K {16..23}+hiA, hiA = (lane>=16)*8.
    // B lane: N = lane&15; same K mapping as A, hiB = (lane>=16)*8.
    const int mrow = lane & 15;
    const int hiA  = (lane >= 16) ? 8 : 0;
    const int hiB  = (lane >= 16) ? 8 : 0;

    const _Float16* arow0 = &Apad[mrow][0];
    const _Float16* arow1 = &Apad[16 + mrow][0];
    const _Float16* brow  = &Bbuf[wave][mrow][0];

    v8f c0 = {}; v8f c1 = {};
    #pragma unroll
    for (int kc = 0; kc < 3; ++kc) {
        const int ka = kc * 32 + hiA;          // 16-B aligned half offsets
        const int kb = kc * 32 + hiB;
        v16h a0 = cat8(*(const v8h*)(arow0 + ka), *(const v8h*)(arow0 + ka + 16));
        v16h a1 = cat8(*(const v8h*)(arow1 + ka), *(const v8h*)(arow1 + ka + 16));
        v16h bf = cat8(*(const v8h*)(brow  + kb), *(const v8h*)(brow  + kb + 16));
        c0 = __builtin_amdgcn_wmma_f32_16x16x32_f16(false, a0, false, bf,
                                                    (short)0, c0, false, false);
        c1 = __builtin_amdgcn_wmma_f32_16x16x32_f16(false, a1, false, bf,
                                                    (short)0, c1, false, false);
    }

    // Scatter C fragments (+bias) to Cbuf[ch][pixel].
    if (lane < 16) {
        #pragma unroll
        for (int r = 0; r < 8; ++r) {
            Cbuf[wave][r][lane]      = c0[r] + b_oa[r];          // ch 0..7
            Cbuf[wave][16 + r][lane] = c1[r] + b_oa[16 + r];     // ch 16..23
        }
    } else {
        #pragma unroll
        for (int r = 0; r < 8; ++r)
            Cbuf[wave][8 + r][lane - 16] = c0[r] + b_oa[8 + r];  // ch 8..15
        // c1 rows map to padded channels 24..31: discarded.
    }
    __syncthreads();

    // TGASS post-processing: lanes 0..15 each own one pixel of the tile.
    if (lane < 16) {
        float C[24];
        #pragma unroll
        for (int ch = 0; ch < 24; ++ch) C[ch] = Cbuf[wave][ch][lane];

        const float scale = aff_scale[0] + 1e-8f;
        float aff[8], s = 1e-4f;
        #pragma unroll
        for (int n = 0; n < 8; ++n) { aff[n] = tanhf(C[16 + n]) / scale; s += fabsf(aff[n]); }
        s = fmaxf(s, 1.0f);
        float inv = 1.0f / s, sum = 0.0f;
        #pragma unroll
        for (int n = 0; n < 8; ++n) { aff[n] *= inv; sum += aff[n]; }
        const float aref = 1.0f - sum;

        const int x = xt + lane;
        const size_t p = ((size_t)(b * HH + y)) * WW + x;
        #pragma unroll
        for (int j = 0; j < 9; ++j) {
            float dy, dx, a;
            if (j == 4) { dy = 0.0f; dx = 0.0f; a = aref; }
            else { int n = (j < 4) ? j : j - 1; dy = C[2 * n]; dx = C[2 * n + 1]; a = aff[n]; }
            float ys = (float)y + (float)(j / 3 - 1) + dy;
            float xs = (float)x + (float)(j % 3 - 1) + dx;
            float y0f = floorf(ys), x0f = floorf(xs);
            int   y0  = (int)y0f,   x0  = (int)x0f;
            size_t o = (size_t)j * NPIX + p;
            tapYX[o] = (((unsigned)(unsigned short)(short)y0) << 16) |
                        ((unsigned)(unsigned short)(short)x0);
            tapF [o] = pack_h2(ys - y0f, xs - x0f);
            tapA [o] = (_Float16)a;
        }
    }
}

// ---------------------------------------------------------------------------
// Kernel 2: g0 = feat_init * conf
// ---------------------------------------------------------------------------
__global__ void __launch_bounds__(256)
init_g_kernel(const float* __restrict__ feat, const float* __restrict__ conf,
              float* __restrict__ g)
{
    size_t p = (size_t)blockIdx.x * 256 + threadIdx.x;
    g[p] = feat[p] * conf[p];
}

// ---------------------------------------------------------------------------
// Kernel 3: one propagation step. Tap state for the block's 256 pixels is
// DMA'd into LDS by the TDM (3 descriptors: 9x256 u32, 9x256 u32, 9x256 f16),
// then each thread does its 9-tap bilinear gather in fp32.
// ---------------------------------------------------------------------------
__global__ void __launch_bounds__(256)
prop_step_kernel(const float* __restrict__ gin,
                 const unsigned* __restrict__ tapYX,
                 const unsigned* __restrict__ tapF,
                 const _Float16* __restrict__ tapA,
                 const float* __restrict__ conf,
                 float* __restrict__ out, int mulConf)
{
    __shared__ unsigned sYX[9][256];   // 9 KB
    __shared__ unsigned sF [9][256];   // 9 KB
    __shared__ _Float16 sA [9][256];   // 4.5 KB

    const int    t    = threadIdx.x;
    const size_t base = (size_t)blockIdx.x * 256;

#if USE_TDM
    if (t == 0) {   // TDM ignores EXEC: one issuing lane -> one DMA per block
        tdm_load_2d((unsigned)(uintptr_t)&sYX[0][0], tapYX + base, 256, 9,
                    (unsigned long long)NPIX, 2u);
        tdm_load_2d((unsigned)(uintptr_t)&sF[0][0],  tapF  + base, 256, 9,
                    (unsigned long long)NPIX, 2u);
        tdm_load_2d((unsigned)(uintptr_t)&sA[0][0],  tapA  + base, 256, 9,
                    (unsigned long long)NPIX, 1u);
    }
    wait_tensorcnt0();          // wave 0 drains its TENSORcnt; no-op elsewhere
    __syncthreads();
#else
    for (int idx = t; idx < 9 * 256; idx += 256) {
        int j = idx >> 8, c = idx & 255;
        size_t o = (size_t)j * NPIX + base + c;
        sYX[j][c] = tapYX[o];
        sF [j][c] = tapF [o];
        sA [j][c] = tapA [o];
    }
    __syncthreads();
#endif

    const size_t p = base + t;
    const int plane = (int)(p / PLANE);
    const float* __restrict__ g = gin + (size_t)plane * PLANE;

    float acc = 0.0f;
    #pragma unroll
    for (int j = 0; j < 9; ++j) {
        unsigned yx = sYX[j][t];
        int y0 = (int)(short)(yx >> 16);
        int x0 = (int)(short)(yx & 0xFFFFu);
        H2U f; f.u = sF[j][t];
        float fy = (float)f.h[0], fx = (float)f.h[1];
        float a  = (float)sA[j][t];

        float wy1 = fy, wy0 = 1.0f - fy;
        float wx1 = fx, wx0 = 1.0f - fx;
        bool xv0 = (x0 >= 0) && (x0 < WW);
        bool xv1 = (x0 + 1 >= 0) && (x0 + 1 < WW);
        float g00 = 0.0f, g01 = 0.0f, g10 = 0.0f, g11 = 0.0f;
        if (y0 >= 0 && y0 < HH) {
            const float* r0 = g + (size_t)y0 * WW;
            if (xv0) g00 = r0[x0];
            if (xv1) g01 = r0[x0 + 1];
        }
        if (y0 + 1 >= 0 && y0 + 1 < HH) {
            const float* r1 = g + (size_t)(y0 + 1) * WW;
            if (xv0) g10 = r1[x0];
            if (xv1) g11 = r1[x0 + 1];
        }
        acc += a * (wy0 * (wx0 * g00 + wx1 * g01) + wy1 * (wx0 * g10 + wx1 * g11));
    }
    out[p] = mulConf ? acc * conf[p] : acc;
}

// ---------------------------------------------------------------------------
// Host launcher. d_ws layout (all sections 256-B aligned since NPIX % 256==0):
//   [0)                tapYX : 9*NPIX*4 = 61,636,608 B
//   [+61,636,608)      tapF  : 9*NPIX*4
//   [+123,273,216)     tapA  : 9*NPIX*2 = 30,818,304 B
//   [+154,091,520)     gA    : NPIX*4
//   [+160,940,032)     gB    : NPIX*4     (total 167,788,544 B)
// ---------------------------------------------------------------------------
extern "C" void kernel_launch(void* const* d_in, const int* in_sizes, int n_in,
                              void* d_out, int out_size, void* d_ws, size_t ws_size,
                              hipStream_t stream)
{
    const float* feat_init  = (const float*)d_in[0];
    const float* guidance   = (const float*)d_in[1];
    const float* confidence = (const float*)d_in[2];
    const float* w_oa       = (const float*)d_in[3];
    const float* b_oa       = (const float*)d_in[4];
    const float* aff_scale  = (const float*)d_in[5];
    float* out = (float*)d_out;

    char* ws = (char*)d_ws;
    const size_t yxBytes = (size_t)9 * NPIX * 4;
    const size_t aBytes  = (size_t)9 * NPIX * 2;
    unsigned* tapYX = (unsigned*)(ws);
    unsigned* tapF  = (unsigned*)(ws + yxBytes);
    _Float16* tapA  = (_Float16*)(ws + 2 * yxBytes);
    float*    gA    = (float*)(ws + 2 * yxBytes + aBytes);
    float*    gB    = gA + NPIX;

    // Phase 1: offsets + affinities via WMMA GEMM.
    offset_aff_wmma_kernel<<<NTILE / 8, 256, 0, stream>>>(
        guidance, w_oa, b_oa, aff_scale, tapYX, tapF, tapA);

    // Phase 2: 18 propagation steps with global sync between launches.
    init_g_kernel<<<NPIX / 256, 256, 0, stream>>>(feat_init, confidence, gA);
    const float* cur = gA;
    float*       nxt = gB;
    for (int t = 0; t < PROP_STEPS; ++t) {
        const bool last = (t == PROP_STEPS - 1);
        prop_step_kernel<<<NPIX / 256, 256, 0, stream>>>(
            cur, tapYX, tapF, tapA, confidence,
            last ? out : nxt, last ? 0 : 1);
        const float* tmp = cur; cur = nxt; nxt = (float*)tmp;
    }
}
